// MultiHeadAttention_20048907337901
// MI455X (gfx1250) — matrix operand order repaired and checked by reading, not compiled, not run on hardware
//
#include <hip/hip_runtime.h>

#define DMODEL 512
#define HEADS  8
#define DK     64
#define BATCH  2
#define SEQ    4096
#define QBLOCK 128

typedef __bf16 bf16_t;
typedef __attribute__((ext_vector_type(16))) __bf16 v16bf;
typedef __attribute__((ext_vector_type(8)))  __bf16 v8bf;
typedef __attribute__((ext_vector_type(8)))  float  v8f;
typedef int v4i_t __attribute__((vector_size(16)));   // matches builtin param

#if defined(__has_builtin)
#if __has_builtin(__builtin_amdgcn_global_load_async_to_lds_b128)
#define HAVE_ASYNC_LDS 1
#endif
#endif

union V16 { v16bf v; v8bf h[2]; };

// A-matrix (16x32, MxK) per-lane layout: lane l row M=l; hi half selects
// K = {hi*8..hi*8+7, 16+hi*8..23+hi*8}  -> two aligned 16B runs.
__device__ __forceinline__ v16bf load_a_frag(const bf16_t* row, int hi) {
  V16 u;
  u.h[0] = *(const v8bf*)(row + hi * 8);
  u.h[1] = *(const v8bf*)(row + 16 + hi * 8);
  return u.v;
}
// B-matrix (32x16, KxN) per-lane layout: lane n column N=n; hi half selects
// K = {hi*8..hi*8+7, 16+hi*8..23+hi*8}  -> two aligned 16B runs.
__device__ __forceinline__ v16bf load_b_frag(const bf16_t* row, int hi) {
  V16 u;
  u.h[0] = *(const v8bf*)(row + hi * 8);
  u.h[1] = *(const v8bf*)(row + 16 + hi * 8);
  return u.v;
}
__device__ __forceinline__ v8f wmma_bf16(v16bf a, v16bf b, v8f c) {
  return __builtin_amdgcn_wmma_f32_16x16x32_bf16(false, a, false, b, (short)0, c,
                                                 false, false);
}

// Stage 16 bytes global -> LDS (async ASYNCcnt-tracked path when available).
__device__ __forceinline__ void stage16(const bf16_t* g, bf16_t* lds_ptr) {
#ifdef HAVE_ASYNC_LDS
  typedef __attribute__((address_space(1))) v4i_t* as1_v4i;
  typedef __attribute__((address_space(3))) v4i_t* as3_v4i;
  __builtin_amdgcn_global_load_async_to_lds_b128(
      (as1_v4i)(size_t)g, (as3_v4i)(size_t)lds_ptr, 0, 0);
#else
  *(v8bf*)lds_ptr = *(const v8bf*)g;
#endif
}
__device__ __forceinline__ void stage_wait() {
#ifdef HAVE_ASYNC_LDS
  asm volatile("s_wait_asynccnt 0" ::: "memory");
#endif
}

// ---------------------------------------------------------------- convert ---
__global__ void cvt_f32_bf16(const float* __restrict__ in, bf16_t* __restrict__ out,
                             int n) {
  int i = blockIdx.x * blockDim.x + threadIdx.x;
  int stride = gridDim.x * blockDim.x;
  for (; i < n; i += stride) out[i] = (bf16_t)in[i];
}

// ----------------------------------------------------------- projections ----
// out = x @ W^T per head. mode 0/1: store [b,h,s,e] (*scale). mode 2: store
// transposed [b,h,e,s] (for V, so PV B-frags are contiguous later).
__global__ void __launch_bounds__(128) proj_kernel(const bf16_t* __restrict__ x,
                                                   const bf16_t* __restrict__ w,
                                                   bf16_t* __restrict__ out,
                                                   int mode, float scale) {
  int wid  = threadIdx.x >> 5;
  int lane = threadIdx.x & 31;
  int l = lane & 15, hi = lane >> 4;
  int id  = blockIdx.x * 4 + wid;          // one 16-row tile per wave
  int st  = id & (SEQ / 16 - 1);
  int tmp = id >> 8;                       // SEQ/16 == 256
  int h = tmp & (HEADS - 1);
  int b = tmp >> 3;
  int s0 = st * 16;

  const bf16_t* xrow  = x + (size_t)(b * SEQ + s0 + l) * DMODEL;
  const bf16_t* wbase = w + (size_t)h * DK * DMODEL;

  v8f c[4] = {{}, {}, {}, {}};
  for (int kb = 0; kb < DMODEL; kb += 32) {
    v16bf a = load_a_frag(xrow + kb, hi);
#pragma unroll
    for (int t = 0; t < 4; ++t) {
      const bf16_t* wrow = wbase + (size_t)(t * 16 + l) * DMODEL + kb;
      c[t] = wmma_bf16(a, load_b_frag(wrow, hi), c[t]);
    }
  }
#pragma unroll
  for (int t = 0; t < 4; ++t)
#pragma unroll
    for (int r = 0; r < 8; ++r) {
      float val = c[t][r] * scale;
      int e = t * 16 + l;
      int s = s0 + r + 8 * hi;
      if (mode < 2)
        out[((size_t)(b * HEADS + h) * SEQ + s) * DK + e] = (bf16_t)val;
      else
        out[((size_t)(b * HEADS + h) * DK + e) * SEQ + s] = (bf16_t)val;
    }
}

// -------------------------------------------------------- flash attention ---
// 256 threads = 8 waves; block owns 128 query rows; K/V^T tiles staged in LDS
// once per 32-key block and shared by all waves (8x less L2 traffic).
__global__ void __launch_bounds__(256) flash_attn_kernel(const bf16_t* __restrict__ qh,
                                                         const bf16_t* __restrict__ kh,
                                                         const bf16_t* __restrict__ vt,
                                                         bf16_t* __restrict__ heads) {
  __shared__ __align__(16) bf16_t kt[32 * 72];        // K tile, padded rows
  __shared__ __align__(16) bf16_t vtt[64 * 40];       // V^T tile, padded rows
  __shared__ __align__(16) bf16_t plds[8][16 * 40];   // per-wave P transpose

  int tid  = threadIdx.x;
  int wid  = tid >> 5;
  int lane = tid & 31;
  int l = lane & 15, hi = lane >> 4;

  int blk = blockIdx.x;
  int qb  = blk & (SEQ / QBLOCK - 1);      // SEQ/QBLOCK == 32
  int tmp = blk >> 5;
  int h = tmp & (HEADS - 1);
  int b = tmp >> 3;
  size_t bh = (size_t)(b * HEADS + h);

  const bf16_t* qbase = qh + bh * SEQ * DK;
  const bf16_t* kbase = kh + bh * SEQ * DK;
  const bf16_t* vbase = vt + bh * DK * SEQ;

  int q0b = qb * QBLOCK;
  int q0  = q0b + wid * 16;                // this wave's query tile

  const bf16_t* qrow = qbase + (size_t)(q0 + l) * DK;
  v16bf qa0 = load_a_frag(qrow, hi);       // K = 0..31  (q pre-scaled 1/8)
  v16bf qa1 = load_a_frag(qrow + 32, hi);  // K = 32..63

  v8f o[4] = {{}, {}, {}, {}};
  float m[8], ls[8];
#pragma unroll
  for (int r = 0; r < 8; ++r) { m[r] = -3.0e38f; ls[r] = 0.0f; }
  bf16_t* pl = &plds[wid][0];

  // cooperative staging coordinates: 256 threads x 16B chunks
  int krow = tid >> 3, koff = (tid & 7) * 8;   // K tile: 32 rows x 64 el
  int vrow = tid >> 2, voff = (tid & 3) * 8;   // V^T tile: 64 rows x 32 el

  int kendw = q0 + 16;                     // causal end for this wave
  int kendb = q0b + QBLOCK;                // causal end for the block

  for (int k0 = 0; k0 < kendb; k0 += 32) {
    __syncthreads();                       // previous-iteration reads done
    const bf16_t* gk = kbase + (size_t)(k0 + krow) * DK + koff;
    const bf16_t* gv = vbase + (size_t)vrow * SEQ + k0 + voff;
    stage16(gk, &kt[krow * 72 + koff]);
    stage16(gv, &vtt[vrow * 40 + voff]);
    if (k0 + 32 < kendb) {
      __builtin_prefetch(gk + 32 * DK, 0, 3);
      __builtin_prefetch(gv + 32, 0, 3);
    }
    stage_wait();                          // drain ASYNCcnt before barrier
    __syncthreads();                       // tiles visible block-wide

    if (k0 < kendw) {                      // wave-uniform causal guard
      // ---- scores: S = Q * K^T, two 16-key N tiles, K over DK ----
      v8f s0 = {}, s1 = {};
      s0 = wmma_bf16(qa0, load_b_frag(&kt[l * 72], hi), s0);
      s0 = wmma_bf16(qa1, load_b_frag(&kt[l * 72 + 32], hi), s0);
      s1 = wmma_bf16(qa0, load_b_frag(&kt[(16 + l) * 72], hi), s1);
      s1 = wmma_bf16(qa1, load_b_frag(&kt[(16 + l) * 72 + 32], hi), s1);

      // ---- causal mask on diagonal blocks ----
      if (k0 + 31 > q0) {
#pragma unroll
        for (int r = 0; r < 8; ++r) {
          int qi = q0 + r + 8 * hi;
          if (k0 + l > qi)      s0[r] = -3.0e38f;
          if (k0 + 16 + l > qi) s1[r] = -3.0e38f;
        }
      }

      // ---- online softmax (rows live in 16-lane halves; shfl_xor <= 8) ----
#pragma unroll
      for (int r = 0; r < 8; ++r) {
        float vmax = fmaxf(s0[r], s1[r]);
        vmax = fmaxf(vmax, __shfl_xor(vmax, 1));
        vmax = fmaxf(vmax, __shfl_xor(vmax, 2));
        vmax = fmaxf(vmax, __shfl_xor(vmax, 4));
        vmax = fmaxf(vmax, __shfl_xor(vmax, 8));
        float mnew = fmaxf(m[r], vmax);
        float sc = __expf(m[r] - mnew);
        m[r] = mnew;
        float p0 = __expf(s0[r] - mnew);
        float p1 = __expf(s1[r] - mnew);
        float rs = p0 + p1;
        rs += __shfl_xor(rs, 1);
        rs += __shfl_xor(rs, 2);
        rs += __shfl_xor(rs, 4);
        rs += __shfl_xor(rs, 8);
        ls[r] = ls[r] * sc + rs;
        o[0][r] *= sc; o[1][r] *= sc; o[2][r] *= sc; o[3][r] *= sc;
        pl[(r + 8 * hi) * 40 + l]      = (bf16_t)p0;   // D-layout -> [M][N]
        pl[(r + 8 * hi) * 40 + 16 + l] = (bf16_t)p1;
      }

      // ---- O += P * V : P as A-frag from LDS, B-frags from V^T tile ----
      v16bf pa = load_a_frag(pl + l * 40, hi);
#pragma unroll
      for (int t = 0; t < 4; ++t)
        o[t] = wmma_bf16(pa, load_b_frag(&vtt[(t * 16 + l) * 40], hi), o[t]);
    }
  }

  float inv[8];
#pragma unroll
  for (int r = 0; r < 8; ++r) inv[r] = 1.0f / ls[r];
#pragma unroll
  for (int t = 0; t < 4; ++t)
#pragma unroll
    for (int r = 0; r < 8; ++r) {
      int s = q0 + r + 8 * hi;
      int col = h * DK + t * 16 + l;
      heads[(size_t)(b * SEQ + s) * DMODEL + col] = (bf16_t)(o[t][r] * inv[r]);
    }
}

// -------------------------------------------------------- output projection -
__global__ void __launch_bounds__(128) out_proj_kernel(const bf16_t* __restrict__ heads,
                                                       const bf16_t* __restrict__ wo,
                                                       float* __restrict__ out) {
  int wid  = threadIdx.x >> 5;
  int lane = threadIdx.x & 31;
  int l = lane & 15, hi = lane >> 4;
  int id = blockIdx.x * 4 + wid;
  int nt = id & 7;                          // 8 groups of 64 output cols
  int rt = id >> 3;
  int r0 = rt * 16;

  const bf16_t* arow = heads + (size_t)(r0 + l) * DMODEL;
  v8f c[4] = {{}, {}, {}, {}};
  for (int kb = 0; kb < DMODEL; kb += 32) {
    v16bf a = load_a_frag(arow + kb, hi);
#pragma unroll
    for (int t = 0; t < 4; ++t) {
      const bf16_t* wrow = wo + (size_t)(nt * 64 + t * 16 + l) * DMODEL + kb;
      c[t] = wmma_bf16(a, load_b_frag(wrow, hi), c[t]);
    }
  }
#pragma unroll
  for (int t = 0; t < 4; ++t)
#pragma unroll
    for (int r = 0; r < 8; ++r)
      out[(size_t)(r0 + r + 8 * hi) * DMODEL + nt * 64 + t * 16 + l] = c[t][r];
}

// -------------------------------------------------------------------- host --
extern "C" void kernel_launch(void* const* d_in, const int* in_sizes, int n_in,
                              void* d_out, int out_size, void* d_ws, size_t ws_size,
                              hipStream_t stream) {
  (void)in_sizes; (void)n_in; (void)out_size; (void)ws_size;
  const float* q  = (const float*)d_in[0];
  const float* k  = (const float*)d_in[1];
  const float* v  = (const float*)d_in[2];
  const float* Wq = (const float*)d_in[3];
  const float* Wk = (const float*)d_in[4];
  const float* Wv = (const float*)d_in[5];
  const float* Wo = (const float*)d_in[6];
  // d_in[7] = mask: causal, handled analytically.
  float* out = (float*)d_out;

  char* ws = (char*)d_ws;
  size_t off = 0;
  auto walloc = [&](size_t elems) {
    bf16_t* p = (bf16_t*)(ws + off);
    off += (elems * sizeof(bf16_t) + 255) & ~(size_t)255;
    return p;
  };
  const size_t NX  = (size_t)BATCH * SEQ * DMODEL;   // 4.19M
  const size_t NWH = (size_t)HEADS * DK * DMODEL;    // 262k
  const size_t NWO = (size_t)DMODEL * DMODEL;        // 262k

  bf16_t* xq  = walloc(NX);
  bf16_t* xk  = walloc(NX);
  bf16_t* xv  = walloc(NX);
  bf16_t* wqb = walloc(NWH);
  bf16_t* wkb = walloc(NWH);
  bf16_t* wvb = walloc(NWH);
  bf16_t* wob = walloc(NWO);
  bf16_t* qhb = walloc(NX);    // [b,h,s,dk], pre-scaled 1/sqrt(DK)
  bf16_t* khb = walloc(NX);    // [b,h,s,dk]
  bf16_t* vtb = walloc(NX);    // [b,h,dk,s]
  bf16_t* hdb = walloc(NX);    // [b,s,h*dk]

  cvt_f32_bf16<<<1024, 256, 0, stream>>>(q,  xq,  (int)NX);
  cvt_f32_bf16<<<1024, 256, 0, stream>>>(k,  xk,  (int)NX);
  cvt_f32_bf16<<<1024, 256, 0, stream>>>(v,  xv,  (int)NX);
  cvt_f32_bf16<<<256,  256, 0, stream>>>(Wq, wqb, (int)NWH);
  cvt_f32_bf16<<<256,  256, 0, stream>>>(Wk, wkb, (int)NWH);
  cvt_f32_bf16<<<256,  256, 0, stream>>>(Wv, wvb, (int)NWH);
  cvt_f32_bf16<<<256,  256, 0, stream>>>(Wo, wob, (int)NWO);

  // projections: 4096 waves each (one 16-row tile/wave), 4 waves/block
  proj_kernel<<<1024, 128, 0, stream>>>(xq, wqb, qhb, 0, 0.125f);
  proj_kernel<<<1024, 128, 0, stream>>>(xk, wkb, khb, 1, 1.0f);
  proj_kernel<<<1024, 128, 0, stream>>>(xv, wvb, vtb, 2, 1.0f);

  // attention: 512 blocks x 8 waves; block owns 128 query rows
  flash_attn_kernel<<<BATCH * HEADS * (SEQ / QBLOCK), 256, 0, stream>>>(qhb, khb,
                                                                        vtb, hdb);

  out_proj_kernel<<<1024, 128, 0, stream>>>(hdb, wob, out);
}
